// LatentCorrelationLearnerNTF_60722247631696
// MI455X (gfx1250) — matrix operand order repaired and checked by reading, not compiled, not run on hardware
//
#include <hip/hip_runtime.h>
#include <stdint.h>

// ---------------- problem constants ----------------
#define B_   8
#define T_   64
#define NN   512          // nodes
#define FF   64           // input features
#define HH   128          // hidden
#define FO   64           // output features
#define BT   512          // B*T
#define BTF  32768        // BT*FF  (K-dim of the cosine-similarity SYRK)
#define NS   3            // num scales
#define KCH  3            // cheb_k
#define TOPK 358          // max(1, int(512*0.7))
#define EPSV 1e-8f

typedef _Float16 h16;
typedef __attribute__((ext_vector_type(16))) _Float16 v16h;
typedef __attribute__((ext_vector_type(8)))  float    v8f;
typedef __attribute__((ext_vector_type(4)))  unsigned int u32x4; // trivially-constructible 16B vector
typedef __attribute__((ext_vector_type(2)))  float    f32x2;

// ---------------- small helpers ----------------
__device__ inline float wave_sum(float v) {
  #pragma unroll
  for (int m = 16; m; m >>= 1) v += __shfl_xor(v, m, 32);
  return v;
}

__device__ inline void softmax3(const float* w, float* o) {
  float a = w[0], b = w[1], c = w[2];
  float m = fmaxf(a, fmaxf(b, c));
  float ea = __expf(a - m), eb = __expf(b - m), ec = __expf(c - m);
  float s = ea + eb + ec;
  o[0] = ea / s; o[1] = eb / s; o[2] = ec / s;
}

__device__ inline float clampf(float v, float lo, float hi) {
  return fminf(hi, fmaxf(lo, v));
}

// ---------------- WMMA fragment loaders (ISA 7.12.2 layouts) ----------------
// A-matrix 16x32 f16: lanes 0-15 -> M=lane, v0..3 = K[k..k+7],  v4..7 = K[k+16..k+23]
//                     lanes16-31 -> M=lane-16, v0..3 = K[k+8..k+15], v4..7 = K[k+24..k+31]
__device__ inline v16h load_a_frag(const h16* A, int lda, int row, int k, int hl) {
  const h16* p = A + (size_t)row * lda + k + 8 * hl;
  union { u32x4 q[2]; v16h v; } u;
  u.q[0] = *reinterpret_cast<const u32x4*>(p);
  u.q[1] = *reinterpret_cast<const u32x4*>(p + 16);
  return u.v;
}
// B-matrix 32x16 f16 (we feed B^T, row-major N x K): lane col = n, K order same as A-matrix
__device__ inline v16h load_b_frag(const h16* Bt, int ldb, int col, int k, int hl) {
  const h16* p = Bt + (size_t)col * ldb + k + 8 * hl;
  union { u32x4 q[2]; v16h v; } u;
  u.q[0] = *reinterpret_cast<const u32x4*>(p);
  u.q[1] = *reinterpret_cast<const u32x4*>(p + 16);
  return u.v;
}

__device__ inline v8f wmma16(v16h a, v16h b, v8f c) {
  return __builtin_amdgcn_wmma_f32_16x16x32_f16(false, a, false, b, (short)0, c,
                                                false, false);
}

// One wave computes a 32x32 tile of C = A(MxK) * B(KxN), with B given transposed
// (Bt: N x K row-major).  K multiple of 32; m0,n0 multiples of 32.
__device__ inline void gemm2x2(const h16* A, int lda, const h16* Bt, int ldb,
                               int K, int m0, int n0, v8f acc[2][2]) {
  const int lane = threadIdx.x & 31;
  const int hl   = lane >> 4;
  const int l16  = lane & 15;
  for (int k = 0; k < K; k += 32) {
    v16h a0 = load_a_frag(A, lda, m0 + l16,      k, hl);
    v16h a1 = load_a_frag(A, lda, m0 + 16 + l16, k, hl);
    v16h b0 = load_b_frag(Bt, ldb, n0 + l16,      k, hl);
    v16h b1 = load_b_frag(Bt, ldb, n0 + 16 + l16, k, hl);
    acc[0][0] = wmma16(a0, b0, acc[0][0]);
    acc[0][1] = wmma16(a0, b1, acc[0][1]);
    acc[1][0] = wmma16(a1, b0, acc[1][0]);
    acc[1][1] = wmma16(a1, b1, acc[1][1]);
  }
}

#define ZERO_ACC(acc)                                         \
  {                                                           \
    v8f z = {0.f, 0.f, 0.f, 0.f, 0.f, 0.f, 0.f, 0.f};         \
    acc[0][0] = z; acc[0][1] = z; acc[1][0] = z; acc[1][1] = z;\
  }

// C/D layout: VGPR j -> (row = m0+16*ti + j + 8*half, col = n0+16*tj + l16)
#define EPILOGUE_LOOP(body)                                   \
  {                                                           \
    const int lane = threadIdx.x & 31;                        \
    const int hl = lane >> 4, l16 = lane & 15;                 \
    _Pragma("unroll") for (int ti = 0; ti < 2; ti++)          \
    _Pragma("unroll") for (int tj = 0; tj < 2; tj++)          \
    _Pragma("unroll") for (int j = 0; j < 8; j++) {           \
      int row = m0 + 16 * ti + j + 8 * hl;                    \
      int col = n0 + 16 * tj + l16;                           \
      float accv = acc[ti][tj][j];                            \
      body                                                    \
    }                                                         \
  }

// ================= graph-structure kernels =================

// A = tanh(0.5*(Ap + Ap^T))
__global__ void k_prep_A(const float* __restrict__ Ap, float* __restrict__ A) {
  int i = blockIdx.x;
  for (int j = threadIdx.x; j < NN; j += blockDim.x)
    A[(size_t)i * NN + j] = tanhf(0.5f * (Ap[(size_t)i * NN + j] + Ap[(size_t)j * NN + i]));
}

// per-row top-k mask via bitonic sort of (value, index); descending, ties -> lower index first
__global__ void k_topk(const float* __restrict__ A, float* __restrict__ Al) {
  __shared__ float sv[NN];
  __shared__ int   si[NN];
  int i = blockIdx.x;
  for (int j = threadIdx.x; j < NN; j += 256) {
    sv[j] = A[(size_t)i * NN + j]; si[j] = j;
    Al[(size_t)i * NN + j] = 0.f;
  }
  __syncthreads();
  for (int sz = 2; sz <= NN; sz <<= 1) {
    for (int st = sz >> 1; st > 0; st >>= 1) {
      for (int t = threadIdx.x; t < NN; t += 256) {
        int p = t ^ st;
        if (p > t) {
          float v0 = sv[t], v1 = sv[p];
          int   i0 = si[t], i1 = si[p];
          bool firstOK = (v0 > v1) || (v0 == v1 && i0 < i1);  // descending order pred
          bool descend = ((t & sz) == 0);
          bool doswap = descend ? !firstOK : firstOK;
          if (doswap) { sv[t] = v1; sv[p] = v0; si[t] = i1; si[p] = i0; }
        }
      }
      __syncthreads();
    }
  }
  for (int t = threadIdx.x; t < TOPK; t += 256)
    Al[(size_t)i * NN + si[t]] = sv[t];
}

// normalize rows of x (center over F, L2-normalize) -> Cb[n][bt*FF+f] (f16, SYRK operand)
__global__ void k_xnorm(const float* __restrict__ x, h16* __restrict__ Cb) {
  int wid = threadIdx.x >> 5, lane = threadIdx.x & 31;
  size_t r = (size_t)blockIdx.x * 8 + wid;   // r in [0, BT*NN)
  int bt = (int)(r / NN), n = (int)(r % NN);
  const float* xr = x + ((size_t)bt * NN + n) * FF;
  f32x2 v = *reinterpret_cast<const f32x2*>(xr + lane * 2);
  float mean = wave_sum(v.x + v.y) * (1.f / FF);
  float c0 = v.x - mean, c1 = v.y - mean;
  float ss = wave_sum(c0 * c0 + c1 * c1);
  float inv = 1.f / fmaxf(sqrtf(ss), EPSV);
  h16* o = Cb + (size_t)n * BTF + (size_t)bt * FF + lane * 2;
  o[0] = (h16)(c0 * inv); o[1] = (h16)(c1 * inv);
}

// A_data = clip( Cb * Cb^T / BT , -1, 1 )   (512x512x32768 SYRK via WMMA)
__global__ void k_gemm_adata(const h16* __restrict__ Cb, float* __restrict__ Adata) {
  int wave = blockIdx.x * 8 + (threadIdx.x >> 5);
  int m0 = (wave / 16) * 32, n0 = (wave % 16) * 32;
  v8f acc[2][2]; ZERO_ACC(acc);
  gemm2x2(Cb, BTF, Cb, BTF, BTF, m0, n0, acc);
  EPILOGUE_LOOP({
    Adata[(size_t)row * NN + col] = clampf(accv * (1.f / BT), -1.f, 1.f);
  })
}

// A0 = (alpha*A_learned + (1-alpha)*A_data) with zero diag;  deg = rowsum(A0)
__global__ void k_blend_deg(const float* __restrict__ Al, const float* __restrict__ Ad,
                            const float* __restrict__ alogit,
                            float* __restrict__ A0, float* __restrict__ deg) {
  __shared__ float red[256];
  int i = blockIdx.x;
  float alpha = 1.f / (1.f + __expf(-alogit[0]));
  float s = 0.f;
  for (int j = threadIdx.x; j < NN; j += 256) {
    float v = alpha * Al[(size_t)i * NN + j] + (1.f - alpha) * Ad[(size_t)i * NN + j];
    if (j == i) v = 0.f;
    A0[(size_t)i * NN + j] = v;
    s += v;
  }
  red[threadIdx.x] = s; __syncthreads();
  for (int st = 128; st > 0; st >>= 1) {
    if (threadIdx.x < st) red[threadIdx.x] += red[threadIdx.x + st];
    __syncthreads();
  }
  if (threadIdx.x == 0) deg[i] = red[0];
}

// L = clip(I - dinv_i*A0*dinv_j, +-1.5) -> f16 row-major (Lb) and transposed (LbT)
__global__ void k_lap(const float* __restrict__ A0, const float* __restrict__ deg,
                      h16* __restrict__ Lb, h16* __restrict__ LbT) {
  int i = blockIdx.x;
  float di = rsqrtf(fmaxf(deg[i], EPSV));
  for (int j = threadIdx.x; j < NN; j += 256) {
    float dj = rsqrtf(fmaxf(deg[j], EPSV));
    float v = ((i == j) ? 1.f : 0.f) - di * A0[(size_t)i * NN + j] * dj;
    v = clampf(v, -1.5f, 1.5f);
    Lb[(size_t)i * NN + j]  = (h16)v;
    LbT[(size_t)j * NN + i] = (h16)v;
  }
}

// M2 = L*L (unclipped, f16 row-major for next GEMM); Ls1^T = clip(M2)^T (f16)
__global__ void k_gemm_LL(const h16* __restrict__ Lb, const h16* __restrict__ LbT,
                          h16* __restrict__ M2b, h16* __restrict__ Ls1T) {
  int wave = blockIdx.x * 8 + (threadIdx.x >> 5);
  int m0 = (wave / 16) * 32, n0 = (wave % 16) * 32;
  v8f acc[2][2]; ZERO_ACC(acc);
  gemm2x2(Lb, NN, LbT, NN, NN, m0, n0, acc);
  EPILOGUE_LOOP({
    M2b[(size_t)row * NN + col]  = (h16)accv;
    Ls1T[(size_t)col * NN + row] = (h16)clampf(accv, -1.5f, 1.5f);
  })
}

// Ls2^T = clip(M2*L)^T
__global__ void k_gemm_M2L(const h16* __restrict__ M2b, const h16* __restrict__ LbT,
                           h16* __restrict__ Ls2T) {
  int wave = blockIdx.x * 8 + (threadIdx.x >> 5);
  int m0 = (wave / 16) * 32, n0 = (wave % 16) * 32;
  v8f acc[2][2]; ZERO_ACC(acc);
  gemm2x2(M2b, NN, LbT, NN, NN, m0, n0, acc);
  EPILOGUE_LOOP({
    Ls2T[(size_t)col * NN + row] = (h16)clampf(accv, -1.5f, 1.5f);
  })
}

// ================= feature pipeline =================

// graph_norm of x over node axis (per bt,f) -> xin f16 [bt][n][f]
__global__ void k_gn_in(const float* __restrict__ x, const float* __restrict__ gw,
                        const float* __restrict__ gb, const float* __restrict__ gms,
                        h16* __restrict__ xin) {
  int bt = blockIdx.x, f = threadIdx.x;   // blockDim = 64
  const float* xp = x + (size_t)bt * NN * FF;
  float s = 0.f, ss = 0.f;
  for (int n = 0; n < NN; n++) { float v = xp[(size_t)n * FF + f]; s += v; ss += v * v; }
  float mean = s * (1.f / NN);
  float var  = ss * (1.f / NN) - mean * mean;
  float inv  = rsqrtf(var + EPSV);
  float msc  = mean * gms[0];
  float w = gw[f], b = gb[f];
  h16* xo = xin + (size_t)bt * NN * FF;
  for (int n = 0; n < NN; n++)
    xo[(size_t)n * FF + f] = (h16)((xp[(size_t)n * FF + f] - msc) * inv * w + b);
}

__global__ void k_cvt_f16(const float* __restrict__ src, h16* __restrict__ dst, int n) {
  int i = blockIdx.x * 256 + threadIdx.x;
  if (i < n) dst[i] = (h16)src[i];
}

// hT[bt][h][n] = in_w(HxF) . xin[bt](NxF)^T + in_b[h]   (stored transposed, f16)
__global__ void k_gemm_inproj(const h16* __restrict__ inw, const h16* __restrict__ xin,
                              const float* __restrict__ inb, h16* __restrict__ hT) {
  int bt = blockIdx.y;
  int wave = blockIdx.x * 8 + (threadIdx.x >> 5);
  int m0 = (wave / 16) * 32, n0 = (wave % 16) * 32;   // tilesM=4(H), tilesN=16(N)
  v8f acc[2][2]; ZERO_ACC(acc);
  gemm2x2(inw, FF, xin + (size_t)bt * NN * FF, FF, FF, m0, n0, acc);
  EPILOGUE_LOOP({
    hT[((size_t)bt * HH + row) * NN + col] = (h16)(accv + inb[row]);
  })
}

// hc = (sum_s sw[s]*w[s][0]) * h       (T0 == h for every scale)
__global__ void k_hc_init(const h16* __restrict__ hT, const float* __restrict__ cheb_w,
                          const float* __restrict__ scale_w, float* __restrict__ hc) {
  float sw[3]; softmax3(scale_w, sw);
  float c0 = 0.f;
  for (int s = 0; s < NS; s++) { float w[3]; softmax3(cheb_w + s * KCH, w); c0 += sw[s] * w[0]; }
  size_t i = (size_t)blockIdx.x * 256 + threadIdx.x;
  hc[i] = c0 * (float)hT[i];
}

// T1 = h . Ls  (transposed form: T1T[f][m] = sum_n hT[f][n]*Ls[n][m]); hc += sw*w1*T1
__global__ void k_gemm_cheb1(const h16* __restrict__ hT, const h16* __restrict__ LsT,
                             const float* __restrict__ cheb_w, const float* __restrict__ scale_w,
                             int s, float* __restrict__ hc, h16* __restrict__ T1) {
  int bt = blockIdx.y;
  int wave = blockIdx.x * 8 + (threadIdx.x >> 5);
  int m0 = (wave / 16) * 32, n0 = (wave % 16) * 32;
  v8f acc[2][2]; ZERO_ACC(acc);
  gemm2x2(hT + (size_t)bt * HH * NN, NN, LsT, NN, NN, m0, n0, acc);
  float sw[3]; softmax3(scale_w, sw);
  float w[3];  softmax3(cheb_w + s * KCH, w);
  float coef = sw[s] * w[1];
  EPILOGUE_LOOP({
    size_t idx = ((size_t)bt * HH + row) * NN + col;
    hc[idx] += coef * accv;
    T1[idx] = (h16)accv;
  })
}

// T2 = clip(2*(T1 . Ls) - T0, +-50);  hc += sw*w2*T2   (T0 == h)
__global__ void k_gemm_cheb2(const h16* __restrict__ T1, const h16* __restrict__ hT,
                             const h16* __restrict__ LsT,
                             const float* __restrict__ cheb_w, const float* __restrict__ scale_w,
                             int s, float* __restrict__ hc) {
  int bt = blockIdx.y;
  int wave = blockIdx.x * 8 + (threadIdx.x >> 5);
  int m0 = (wave / 16) * 32, n0 = (wave % 16) * 32;
  v8f acc[2][2]; ZERO_ACC(acc);
  gemm2x2(T1 + (size_t)bt * HH * NN, NN, LsT, NN, NN, m0, n0, acc);
  float sw[3]; softmax3(scale_w, sw);
  float w[3];  softmax3(cheb_w + s * KCH, w);
  float coef = sw[s] * w[2];
  EPILOGUE_LOOP({
    size_t idx = ((size_t)bt * HH + row) * NN + col;
    float t0 = (float)hT[idx];
    hc[idx] += coef * clampf(2.f * accv - t0, -50.f, 50.f);
  })
}

// gelu(graph_norm(hc)) over node axis (hcT layout makes node axis contiguous) -> hh[bt][n][h] f16
__global__ void k_gn_h(const float* __restrict__ hc, const float* __restrict__ gw,
                       const float* __restrict__ gb, const float* __restrict__ gms,
                       h16* __restrict__ hh) {
  int bt = blockIdx.x;
  int wid = threadIdx.x >> 5, lane = threadIdx.x & 31;
  const float* base = hc + (size_t)bt * HH * NN;
  h16* out = hh + (size_t)bt * NN * HH;
  for (int h = wid; h < HH; h += 8) {
    const float* r = base + (size_t)h * NN;
    float s = 0.f, ss = 0.f;
    for (int n = lane; n < NN; n += 32) { float v = r[n]; s += v; ss += v * v; }
    s = wave_sum(s); ss = wave_sum(ss);
    float mean = s * (1.f / NN);
    float var  = ss * (1.f / NN) - mean * mean;
    float inv  = rsqrtf(var + EPSV);
    float msc  = mean * gms[0];
    float w = gw[h], b = gb[h];
    for (int n = lane; n < NN; n += 32) {
      float v = (r[n] - msc) * inv * w + b;
      float u = 0.7978845608f * (v + 0.044715f * v * v * v);
      out[(size_t)n * HH + h] = (h16)(0.5f * v * (1.f + tanhf(u)));
    }
  }
}

// yT[bt][o][n] = out_w(FOxH) . hh[bt](NxH)^T + out_b[o]
__global__ void k_gemm_outproj(const h16* __restrict__ outw, const h16* __restrict__ hh,
                               const float* __restrict__ outb, float* __restrict__ yT) {
  int bt = blockIdx.y;
  int wave = blockIdx.x * 8 + (threadIdx.x >> 5);
  int m0 = (wave / 16) * 32, n0 = (wave % 16) * 32;   // tilesM=2(FO), tilesN=16(N)
  v8f acc[2][2]; ZERO_ACC(acc);
  gemm2x2(outw, HH, hh + (size_t)bt * NN * HH, HH, HH, m0, n0, acc);
  EPILOGUE_LOOP({
    yT[((size_t)bt * FO + row) * NN + col] = accv + outb[row];
  })
}

// final graph_norm over node axis -> d_out[bt][n][o] f32
__global__ void k_gn_out(const float* __restrict__ yT, const float* __restrict__ gw,
                         const float* __restrict__ gb, const float* __restrict__ gms,
                         float* __restrict__ out) {
  int bt = blockIdx.x;
  int wid = threadIdx.x >> 5, lane = threadIdx.x & 31;
  for (int o = wid; o < FO; o += 8) {
    const float* r = yT + ((size_t)bt * FO + o) * NN;
    float s = 0.f, ss = 0.f;
    for (int n = lane; n < NN; n += 32) { float v = r[n]; s += v; ss += v * v; }
    s = wave_sum(s); ss = wave_sum(ss);
    float mean = s * (1.f / NN);
    float var  = ss * (1.f / NN) - mean * mean;
    float inv  = rsqrtf(var + EPSV);
    float msc  = mean * gms[0];
    float w = gw[o], b = gb[o];
    for (int n = lane; n < NN; n += 32)
      out[((size_t)bt * NN + n) * FO + o] = (r[n] - msc) * inv * w + b;
  }
}

// ================= host launch =================
extern "C" void kernel_launch(void* const* d_in, const int* in_sizes, int n_in,
                              void* d_out, int out_size, void* d_ws, size_t ws_size,
                              hipStream_t stream) {
  (void)in_sizes; (void)n_in; (void)out_size; (void)ws_size;
  const float* x       = (const float*)d_in[0];
  const float* A_param = (const float*)d_in[1];
  const float* alogit  = (const float*)d_in[2];
  const float* in_w    = (const float*)d_in[3];
  const float* in_b    = (const float*)d_in[4];
  const float* out_w   = (const float*)d_in[5];
  const float* out_b   = (const float*)d_in[6];
  const float* gin_w   = (const float*)d_in[7];
  const float* gin_b   = (const float*)d_in[8];
  const float* gin_ms  = (const float*)d_in[9];
  const float* gh_w    = (const float*)d_in[10];
  const float* gh_b    = (const float*)d_in[11];
  const float* gh_ms   = (const float*)d_in[12];
  const float* go_w    = (const float*)d_in[13];
  const float* go_b    = (const float*)d_in[14];
  const float* go_ms   = (const float*)d_in[15];
  const float* cheb_w  = (const float*)d_in[16];
  const float* scale_w = (const float*)d_in[17];

  char* cur = (char*)d_ws;
  auto carve = [&](size_t bytes) -> void* {
    void* p = cur;
    cur += (bytes + 255) & ~(size_t)255;
    return p;
  };
  float* A     = (float*)carve((size_t)NN * NN * 4);
  float* Al    = (float*)carve((size_t)NN * NN * 4);
  float* Adata = (float*)carve((size_t)NN * NN * 4);
  float* A0    = (float*)carve((size_t)NN * NN * 4);
  float* deg   = (float*)carve((size_t)NN * 4);
  h16*   Lb    = (h16*)carve((size_t)NN * NN * 2);
  h16*   LbT   = (h16*)carve((size_t)NN * NN * 2);
  h16*   M2b   = (h16*)carve((size_t)NN * NN * 2);
  h16*   Ls1T  = (h16*)carve((size_t)NN * NN * 2);
  h16*   Ls2T  = (h16*)carve((size_t)NN * NN * 2);
  h16*   inw_h = (h16*)carve((size_t)HH * FF * 2);
  h16*   outw_h= (h16*)carve((size_t)FO * HH * 2);
  h16*   Cb    = (h16*)carve((size_t)NN * BTF * 2);
  h16*   xin   = (h16*)carve((size_t)BT * NN * FF * 2);
  h16*   hT    = (h16*)carve((size_t)BT * HH * NN * 2);
  h16*   T1    = (h16*)carve((size_t)BT * HH * NN * 2);
  float* hc    = (float*)carve((size_t)BT * HH * NN * 4);
  h16*   hh    = (h16*)carve((size_t)BT * NN * HH * 2);
  float* yT    = (float*)carve((size_t)BT * FO * NN * 4);

  // ---- graph structure ----
  k_prep_A<<<NN, 256, 0, stream>>>(A_param, A);
  k_topk<<<NN, 256, 0, stream>>>(A, Al);
  k_xnorm<<<BT * NN / 8, 256, 0, stream>>>(x, Cb);
  k_gemm_adata<<<32, 256, 0, stream>>>(Cb, Adata);                 // 256 tiles / 8 waves
  k_blend_deg<<<NN, 256, 0, stream>>>(Al, Adata, alogit, A0, deg);
  k_lap<<<NN, 256, 0, stream>>>(A0, deg, Lb, LbT);
  k_gemm_LL<<<32, 256, 0, stream>>>(Lb, LbT, M2b, Ls1T);
  k_gemm_M2L<<<32, 256, 0, stream>>>(M2b, LbT, Ls2T);

  // ---- feature pipeline ----
  k_gn_in<<<BT, 64, 0, stream>>>(x, gin_w, gin_b, gin_ms, xin);
  k_cvt_f16<<<(HH * FF + 255) / 256, 256, 0, stream>>>(in_w, inw_h, HH * FF);
  k_cvt_f16<<<(FO * HH + 255) / 256, 256, 0, stream>>>(out_w, outw_h, FO * HH);
  k_gemm_inproj<<<dim3(8, BT), 256, 0, stream>>>(inw_h, xin, in_b, hT);   // 64 tiles/bt
  k_hc_init<<<(int)((size_t)BT * HH * NN / 256), 256, 0, stream>>>(hT, cheb_w, scale_w, hc);

  const h16* LsT[3] = {LbT, Ls1T, Ls2T};
  for (int s = 0; s < NS; s++) {
    k_gemm_cheb1<<<dim3(8, BT), 256, 0, stream>>>(hT, LsT[s], cheb_w, scale_w, s, hc, T1);
    k_gemm_cheb2<<<dim3(8, BT), 256, 0, stream>>>(T1, hT, LsT[s], cheb_w, scale_w, s, hc);
  }

  k_gn_h<<<BT, 256, 0, stream>>>(hc, gh_w, gh_b, gh_ms, hh);
  k_gemm_outproj<<<dim3(4, BT), 256, 0, stream>>>(outw_h, hh, out_b, yT); // 32 tiles/bt
  k_gn_out<<<BT, 256, 0, stream>>>(yT, go_w, go_b, go_ms, (float*)d_out);
}
